// CustomSAGE_66348654789006
// MI455X (gfx1250) — matrix operand order repaired and checked by reading, not compiled, not run on hardware
//
#include <hip/hip_runtime.h>
#include <hip/hip_bf16.h>
#include <stdint.h>

// ---------------- problem constants ----------------
#define Nn 30000
#define Ee 960000
#define Hh 300
#define Vv 10000
#define Ll 3
#define HP 320      // H padded to multiple of 32 (K dim for WMMA)
#define MP 30080    // N nodes padded to multiple of 128 (235 tiles)
#define LDW 384     // layer-weight col pad (24 tiles of 16)
#define LDWL 10112  // Wlast col pad (632 tiles of 16)

// ---------------- GEMM tiling ----------------
#define BM 128
#define BN 128
#define BK 32
#define KT10 (HP / BK)  // 10 K tiles, compile-time
#define LDA_S 40        // LDS A row stride in ushort (80B, 16B-aligned, conflict-skewed)

typedef __attribute__((ext_vector_type(16))) __bf16 v16bf;
typedef __attribute__((ext_vector_type(8)))  float  v8f;

union frag16 { uint4 u[2]; v16bf v; };

static __device__ __forceinline__ unsigned short f2bf(float f) {
    unsigned int u = __float_as_uint(f);
    return (unsigned short)((u + 0x7fffu + ((u >> 16) & 1u)) >> 16);  // RNE
}

// ---------------- utility kernels ----------------
__global__ void k_zero_f32(float* __restrict__ p, long n) {
    long t = (long)blockIdx.x * blockDim.x + threadIdx.x;
    if (t < n) p[t] = 0.0f;
}

__global__ void k_deg(const int* __restrict__ dst, float* __restrict__ deg, int e) {
    int t = blockIdx.x * blockDim.x + threadIdx.x;
    if (t < e)
        __hip_atomic_fetch_add(&deg[dst[t]], 1.0f, __ATOMIC_RELAXED, __HIP_MEMORY_SCOPE_AGENT);
}

__global__ void k_invdeg(float* __restrict__ deg, int n) {
    int t = blockIdx.x * blockDim.x + threadIdx.x;
    if (t < n) { float d = deg[t]; deg[t] = (d > 0.0f) ? (1.0f / d) : 0.0f; }
}

__global__ void k_embed(const int* __restrict__ x, const float* __restrict__ emb,
                        float* __restrict__ h) {
    long t = (long)blockIdx.x * blockDim.x + threadIdx.x;
    if (t < (long)Nn * Hh) {
        int n = (int)(t / Hh);
        int j = (int)(t - (long)n * Hh);
        h[t] = emb[(long)x[n] * Hh + j];
    }
}

__global__ void k_scatter(const int* __restrict__ src, const int* __restrict__ dst,
                          const float* __restrict__ h, float* __restrict__ agg) {
    long t = (long)blockIdx.x * blockDim.x + threadIdx.x;
    if (t < (long)Ee * Hh) {
        int e = (int)(t / Hh);
        int j = (int)(t - (long)e * Hh);
        float v = h[(long)src[e] * Hh + j];
        __hip_atomic_fetch_add(&agg[(long)dst[e] * Hh + j], v,
                               __ATOMIC_RELAXED, __HIP_MEMORY_SCOPE_AGENT);
    }
}

// activations: f32 [rows x cols] -> bf16 [Rp x Cp] row-major, zero padded;
// optional per-row scale (inverse degree for the scatter-mean)
__global__ void k_cvt_bf16(const float* __restrict__ srcM, const float* __restrict__ rowScale,
                           unsigned short* __restrict__ dstM,
                           int rows, int cols, int Rp, int Cp) {
    long t = (long)blockIdx.x * blockDim.x + threadIdx.x;
    if (t < (long)Rp * Cp) {
        int r = (int)(t / Cp);
        int c = (int)(t - (long)r * Cp);
        unsigned short o = 0;
        if (r < rows && c < cols) {
            float v = srcM[(long)r * cols + c];
            if (rowScale) v *= rowScale[r];
            o = f2bf(v);
        }
        dstM[t] = o;
    }
}

// weights: f32 [rows x cols] -> bf16 pre-swizzled into WMMA B-fragment layout:
//   dst[((kt*NT + nt)*32 + lane)*16 + e], lane = (c%16) + 16*((k%32)/16), e = k%16
// so each lane's 16x16-col fragment slice is 32 contiguous bytes (2x b128).
__global__ void k_cvt_wswz(const float* __restrict__ W, unsigned short* __restrict__ dstM,
                           int rows, int cols, int Ktiles, int NT) {
    long t = (long)blockIdx.x * blockDim.x + threadIdx.x;
    long total = (long)Ktiles * NT * 512;
    if (t < total) {
        int e    = (int)(t & 15);
        int lane = (int)((t >> 4) & 31);
        int nt   = (int)((t >> 9) % NT);
        int kt   = (int)(t / ((long)NT << 9));
        int k = kt * 32 + (lane >> 4) * 16 + e;
        int c = nt * 16 + (lane & 15);
        unsigned short o = 0;
        if (k < rows && c < cols) o = f2bf(W[(long)k * cols + c]);
        dstM[t] = o;
    }
}

// ---------------- WMMA GEMM ----------------
// C[M x Nreal] (f32, stride ldc) = act( A*Bsw (+ A2*B2sw) + bias )
// A,A2: bf16 [>=gridX*128 x lda] row-major zero-padded.
// Bsw,B2sw: bf16 pre-swizzled B-fragment layout, NT column tiles, NK K tiles.
// A tile streamed into a double-buffered LDS tile with
// GLOBAL_LOAD_ASYNC_TO_LDS_B128 (ASYNCcnt): issue tile k+1, wait for tile k's
// 2 in-order copies (s_wait_asynccnt 2), compute tile k under the async copy.
// NK is compile-time so the pass/kt/buffer bookkeeping folds to constants.
template <bool TWO, bool RELU, int NK>
__global__ __launch_bounds__(256) void k_gemm_bf16(
    const unsigned short* __restrict__ A, const unsigned short* __restrict__ Bsw,
    const unsigned short* __restrict__ A2, const unsigned short* __restrict__ B2sw,
    const float* __restrict__ bias, float* __restrict__ C,
    int M, int Nreal, int ldc, int lda, int NT) {

    __shared__ __align__(16) unsigned short lA[2][BM * LDA_S];

    const int tid  = threadIdx.x;
    const int lane = tid & 31;
    const int w    = tid >> 5;
    const int wm   = w >> 2;   // 0..1 -> 64-row strip
    const int wn   = w & 3;    // 0..3 -> 32-col strip
    const int bm   = blockIdx.x * BM;
    const int bnT  = blockIdx.y * (BN / 16);  // base column tile

    const int lrow = lane & 15;
    const int lkh  = lane >> 4;

    const v8f vzero = {0.f, 0.f, 0.f, 0.f, 0.f, 0.f, 0.f, 0.f};
    v8f acc[4][2];
#pragma unroll
    for (int i = 0; i < 4; ++i)
#pragma unroll
        for (int j = 0; j < 2; ++j) acc[i][j] = vzero;

    constexpr int nPass = TWO ? 2 : 1;

    // per-thread A-staging coordinates: thread copies 32B of one row (2x b128)
    const int srow  = tid >> 1;
    const int shalf = tid & 1;
    const unsigned short* ArowG  = A + (size_t)(bm + srow) * lda + shalf * 16;
    const unsigned short* A2rowG = TWO ? (A2 + (size_t)(bm + srow) * lda + shalf * 16) : A;
    unsigned short* lrowp = &lA[0][srow * LDA_S + shalf * 16];

    auto stage_async = [&](const unsigned short* g, int b) {
        uint32_t loff = (uint32_t)(uintptr_t)(lrowp + b * (BM * LDA_S));
        // INST_OFFSET is added to both LDS and global address for async loads
        asm volatile("global_load_async_to_lds_b128 %0, %1, off"
                     :: "v"(loff), "v"(g) : "memory");
        asm volatile("global_load_async_to_lds_b128 %0, %1, off offset:16"
                     :: "v"(loff), "v"(g) : "memory");
    };

    stage_async(ArowG, 0);  // preload pass 0, kt 0 into buffer 0

#pragma unroll
    for (int pass = 0; pass < nPass; ++pass) {
        const unsigned short* ArG = (TWO && pass) ? A2rowG : ArowG;
        const unsigned short* Bg  = (TWO && pass) ? B2sw : Bsw;
#pragma unroll
        for (int kt = 0; kt < NK; ++kt) {
            const int buf = (pass * NK + kt) & 1;  // compile-time after unroll

            __syncthreads();  // everyone done computing from lA[buf^1]
            if (kt + 1 < NK) {
                stage_async(ArG + (kt + 1) * BK, buf ^ 1);
                asm volatile("s_wait_asynccnt 0x2" ::: "memory");
            } else if (pass + 1 < nPass) {
                stage_async(A2rowG, buf ^ 1);
                asm volatile("s_wait_asynccnt 0x2" ::: "memory");
            } else {
                asm volatile("s_wait_asynccnt 0x0" ::: "memory");
            }
            __syncthreads();  // tile (pass,kt) fully resident in lA[buf]

            // A fragments from LDS: two contiguous 16B runs per lane
            const unsigned short* lab = lA[buf];
            frag16 af[4];
#pragma unroll
            for (int ms = 0; ms < 4; ++ms) {
                const unsigned short* ap =
                    lab + (wm * 64 + ms * 16 + lrow) * LDA_S + lkh * 16;
                af[ms].u[0] = *(const uint4*)(ap);
                af[ms].u[1] = *(const uint4*)(ap + 8);
            }
#pragma unroll
            for (int ns = 0; ns < 2; ++ns) {
                // B fragment: 32 contiguous bytes per lane, pre-swizzled weights
                const int tcol = bnT + wn * 2 + ns;
                const uint4* bp =
                    (const uint4*)(Bg + (((size_t)kt * NT + tcol) * 32 + lane) * 16);
                frag16 bfv;
                bfv.u[0] = bp[0];
                bfv.u[1] = bp[1];
                if (ns == 0 && kt + 1 < NK)
                    __builtin_prefetch(
                        Bg + (((size_t)(kt + 1) * NT + tcol) * 32 + lane) * 16, 0, 0);
#pragma unroll
                for (int ms = 0; ms < 4; ++ms) {
                    acc[ms][ns] = __builtin_amdgcn_wmma_f32_16x16x32_bf16(
                        false, af[ms].v, false, bfv.v, (short)0, acc[ms][ns], false, false);
                }
            }
        }
    }

    // epilogue: C/D layout -> lane n = lane&15, VGPR r -> M = r + 8*(lane>>4)
    const int mh = lane >> 4;
#pragma unroll
    for (int ns = 0; ns < 2; ++ns) {
        int gn = bnT * 16 + wn * 32 + ns * 16 + lrow;
        float bv = (gn < Nreal) ? bias[gn] : 0.0f;
#pragma unroll
        for (int ms = 0; ms < 4; ++ms) {
#pragma unroll
            for (int r = 0; r < 8; ++r) {
                int gm = bm + wm * 64 + ms * 16 + mh * 8 + r;
                if (gm < M && gn < Nreal) {
                    float v = acc[ms][ns][r] + bv;
                    if (RELU) v = fmaxf(v, 0.0f);
                    C[(long)gm * ldc + gn] = v;
                }
            }
        }
    }
}

// ---------------- host launcher ----------------
static inline unsigned blk(long total) { return (unsigned)((total + 255) / 256); }

extern "C" void kernel_launch(void* const* d_in, const int* in_sizes, int n_in,
                              void* d_out, int out_size, void* d_ws, size_t ws_size,
                              hipStream_t stream) {
    const int*   x     = (const int*)d_in[0];
    const int*   ei    = (const int*)d_in[1];
    const float* emb   = (const float*)d_in[2];
    const float* Wl    = (const float*)d_in[3];
    const float* bl    = (const float*)d_in[4];
    const float* Wr    = (const float*)d_in[5];
    const float* Wlast = (const float*)d_in[6];
    const float* blast = (const float*)d_in[7];
    float*       out   = (float*)d_out;

    const int* src = ei;       // edge_index[0]
    const int* dst = ei + Ee;  // edge_index[1]

    // workspace carve-up (256B aligned)
    char*  ws  = (char*)d_ws;
    size_t off = 0;
    auto carve = [&](size_t bytes) {
        size_t o = off; off += (bytes + 255) & ~(size_t)255; return o;
    };
    float*          deg      = (float*)(ws + carve((size_t)Nn * 4));
    float*          h        = (float*)(ws + carve((size_t)Nn * Hh * 4));
    float*          agg      = (float*)(ws + carve((size_t)Nn * Hh * 4));
    unsigned short* h_bf     = (unsigned short*)(ws + carve((size_t)MP * HP * 2));
    unsigned short* agg_bf   = (unsigned short*)(ws + carve((size_t)MP * HP * 2));
    unsigned short* wl_bf    = (unsigned short*)(ws + carve((size_t)HP * LDW * 2));
    unsigned short* wr_bf    = (unsigned short*)(ws + carve((size_t)HP * LDW * 2));
    unsigned short* wlast_bf = (unsigned short*)(ws + carve((size_t)HP * LDWL * 2));
    (void)ws_size; (void)in_sizes; (void)n_in; (void)out_size;

    const int NT_W  = LDW / 16;      // 24 col tiles (layer weights)
    const int NT_WL = LDWL / 16;     // 632 col tiles (Wlast)

    // degree -> inverse degree (in place)
    k_zero_f32<<<blk(Nn), 256, 0, stream>>>(deg, Nn);
    k_deg<<<blk(Ee), 256, 0, stream>>>(dst, deg, Ee);
    k_invdeg<<<blk(Nn), 256, 0, stream>>>(deg, Nn);

    // embedding gather
    k_embed<<<blk((long)Nn * Hh), 256, 0, stream>>>(x, emb, h);

    // final weight -> bf16 swizzled fragments (done once)
    k_cvt_wswz<<<blk((long)KT10 * NT_WL * 512), 256, 0, stream>>>(
        Wlast, wlast_bf, Hh, Vv, KT10, NT_WL);

    for (int i = 0; i < Ll; ++i) {
        // scatter-mean of neighbor features
        k_zero_f32<<<blk((long)Nn * Hh), 256, 0, stream>>>(agg, (long)Nn * Hh);
        k_scatter<<<blk((long)Ee * Hh), 256, 0, stream>>>(src, dst, h, agg);

        // bf16 staging (agg scaled by inv-degree)
        k_cvt_bf16<<<blk((long)MP * HP), 256, 0, stream>>>(agg, deg, agg_bf, Nn, Hh, MP, HP);
        k_cvt_bf16<<<blk((long)MP * HP), 256, 0, stream>>>(h, nullptr, h_bf, Nn, Hh, MP, HP);
        k_cvt_wswz<<<blk((long)KT10 * NT_W * 512), 256, 0, stream>>>(
            Wl + (size_t)i * Hh * Hh, wl_bf, Hh, Hh, KT10, NT_W);
        k_cvt_wswz<<<blk((long)KT10 * NT_W * 512), 256, 0, stream>>>(
            Wr + (size_t)i * Hh * Hh, wr_bf, Hh, Hh, KT10, NT_W);

        // h = relu(agg@Wl + bl + h@Wr)  (fused two-pass WMMA, f32 out into h)
        dim3 g(MP / BM, (Hh + BN - 1) / BN);
        k_gemm_bf16<true, true, KT10><<<g, 256, 0, stream>>>(
            agg_bf, wl_bf, h_bf, wr_bf, bl + (size_t)i * Hh, h,
            Nn, Hh, Hh, HP, NT_W);
    }

    // logits = h @ Wlast + blast
    k_cvt_bf16<<<blk((long)MP * HP), 256, 0, stream>>>(h, nullptr, h_bf, Nn, Hh, MP, HP);
    dim3 gf(MP / BM, (Vv + BN - 1) / BN);
    k_gemm_bf16<false, false, KT10><<<gf, 256, 0, stream>>>(
        h_bf, wlast_bf, nullptr, nullptr, blast, out,
        Nn, Vv, Vv, HP, NT_WL);
}
